// CircularRelativePositionAttention_1271310320141
// MI455X (gfx1250) — matrix operand order repaired and checked by reading, not compiled, not run on hardware
//
#include <hip/hip_runtime.h>
#include <hip/hip_bf16.h>

// ---------------------------------------------------------------------------
// Circular relative-position attention, CDNA5 (gfx1250), bf16 WMMA + fp32 acc.
//   B=4, H=16, S=1024, dh=64.  One block = 32 q-rows of one (b,h).
//   scores (fp32, 32x1024) live entirely in LDS; rel terms become GEMMs:
//     P = Q @ rel_k^T  scattered on circular diagonals k=(q±l)%S
//     out += (attn @ V) + (W @ rel_v),  W[q,l] = attn[q,(q+l)%S]+attn[q,(q-l)%S]
// ---------------------------------------------------------------------------

#define S_LEN   1024
#define NHEADS  16
#define DH      64
#define DMODEL  1024
#define QT      32        // q rows per block
#define LMAX    512       // max circular distance
#define WCOLS   544       // W K-extent padded to 17*32

typedef __attribute__((ext_vector_type(16))) __bf16 v16bf;
typedef __attribute__((ext_vector_type(8)))  float  v8f;

// LDS element strides, padded for bank-conflict-free fragment loads
#define SA_STRIDE 1032    // attn bf16 (1024 + 8)
#define SW_STRIDE 552     // W bf16    (544 + 8)
#define STG_NK    72      // staging [n][k], 64 K + pad
#define STG_KN    136     // staging [n][k], 128 K + pad

#define SS_BYTES  (QT * S_LEN * 4)        // 131072
#define SA_BYTES  (QT * SA_STRIDE * 2)    //  66048
#define SW_BYTES  (QT * SW_STRIDE * 2)    //  35328
#define STG_BYTES (9216 * 2)              //  18432
#define SMEM_BYTES (SS_BYTES + SA_BYTES + SW_BYTES + STG_BYTES)

__device__ __forceinline__ v8f wmma_bf16(v16bf a, v16bf b, v8f c) {
  return __builtin_amdgcn_wmma_f32_16x16x32_bf16(
      /*neg_a=*/false, a, /*neg_b=*/false, b,
      /*c_mod=*/(short)0, c, /*reuse_a=*/false, /*reuse_b=*/false);
}

// 16-bit A fragment (16x32, MxK), ISA 7.12.2 layout:
//  lanes 0-15: M=lane, K in {0..7,16..23}; lanes 16-31: M=lane-16, K in {8..15,24..31}
__device__ __forceinline__ v16bf load_A16(const __bf16* base, int row0, int k0, int stride) {
  const int lane = threadIdx.x & 31;
  const int m    = lane & 15;
  const int klo  = (lane & 16) ? 8 : 0;
  const __bf16* p = base + (row0 + m) * stride + k0 + klo;
  v16bf a;
#pragma unroll
  for (int r = 0; r < 4; ++r) { a[2*r]   = p[2*r];      a[2*r+1]   = p[2*r+1]; }
#pragma unroll
  for (int r = 0; r < 4; ++r) { a[8+2*r] = p[16+2*r];   a[8+2*r+1] = p[16+2*r+1]; }
  return a;
}

// 16-bit B fragment (32x16, KxN) from an LDS tile stored N-major with K contiguous:
//  lanes 0-15: N=lane, K in {0..7,16..23}; lanes 16-31: N=lane-16, K in {8..15,24..31}
__device__ __forceinline__ v16bf load_B16(const __bf16* base, int n0, int k0, int stride) {
  const int lane = threadIdx.x & 31;
  const int n    = lane & 15;
  const int kb   = (lane & 16) ? 8 : 0;
  const __bf16* p = base + (n0 + n) * stride + k0 + kb;
  v16bf b;
#pragma unroll
  for (int r = 0; r < 8; ++r) { const int o = (r < 4) ? 2*r : 16 + 2*(r-4); b[2*r] = p[o]; b[2*r+1] = p[o+1]; }
  return b;
}

__global__ __launch_bounds__(128, 1)
void crpa_wmma_kernel(const float* __restrict__ qg_,
                      const float* __restrict__ kg_,
                      const float* __restrict__ vg_,
                      const float* __restrict__ rk_,
                      const float* __restrict__ rv_,
                      float* __restrict__ out_)
{
  extern __shared__ char smem[];
  float*  sS = (float*)(smem);                                  // [QT][1024] fp32 scores
  __bf16* sA = (__bf16*)(smem + SS_BYTES);                      // [QT][SA_STRIDE] attn bf16
  __bf16* sW = (__bf16*)(smem + SS_BYTES + SA_BYTES);           // [QT][SW_STRIDE] W bf16
  __bf16* sT = (__bf16*)(smem + SS_BYTES + SA_BYTES + SW_BYTES);// staging

  const int tid     = threadIdx.x;
  const int lane    = tid & 31;
  const int wave    = tid >> 5;
  const int rowHalf = wave & 1;   // which 16 q-rows this wave owns
  const int colHalf = wave >> 1;  // which column half this wave owns

  const int nQT   = S_LEN / QT;           // 32
  const int bh    = blockIdx.x / nQT;
  const int qt    = blockIdx.x % nQT;
  const int b     = bh / NHEADS;
  const int h     = bh % NHEADS;
  const int qBase = qt * QT;

  const float* qbh = qg_ + (size_t)b * S_LEN * DMODEL + h * DH;
  const float* kbh = kg_ + (size_t)b * S_LEN * DMODEL + h * DH;
  const float* vbh = vg_ + (size_t)b * S_LEN * DMODEL + h * DH;
  float*       obh = out_ + (size_t)b * S_LEN * DMODEL + h * DH;

  // ---- Stage Q tile (QT x DH) -> bf16, hoist per-wave A fragments to VGPRs
  for (int e = tid; e < QT * DH; e += 128) {
    int r = e >> 6, c = e & 63;
    sT[r * STG_NK + c] = (__bf16)qbh[(size_t)(qBase + r) * DMODEL + c];
  }
  __syncthreads();
  v16bf aq0 = load_A16(sT, rowHalf * 16, 0,  STG_NK);
  v16bf aq1 = load_A16(sT, rowHalf * 16, 32, STG_NK);

  // ---- Step 1: scores = (Q K^T) / sqrt(dh), key cols streamed in 8x128 chunks
  for (int c = 0; c < 8; ++c) {
    __syncthreads();
    for (int e = tid; e < 128 * DH; e += 128) {
      int r = e >> 6, cc = e & 63;
      sT[r * STG_NK + cc] = (__bf16)kbh[(size_t)(c * 128 + r) * DMODEL + cc];
    }
    __syncthreads();
#pragma unroll
    for (int nt = 0; nt < 4; ++nt) {
      const int n0 = colHalf * 64 + nt * 16;
      v8f acc = {};
      acc = wmma_bf16(aq0, load_B16(sT, n0, 0,  STG_NK), acc);
      acc = wmma_bf16(aq1, load_B16(sT, n0, 32, STG_NK), acc);
      const int row0 = rowHalf * 16 + ((lane & 16) ? 8 : 0);
      const int col  = c * 128 + n0 + (lane & 15);
#pragma unroll
      for (int r = 0; r < 8; ++r)
        sS[(row0 + r) * S_LEN + col] = acc[r] * 0.125f;   // 1/sqrt(64)
    }
  }

  // ---- Step 2: scores += Q·rel_k[idx] via P = Q rel_k^T, diagonal scatter
  for (int c = 0; c < 5; ++c) {
    __syncthreads();
    for (int e = tid; e < 128 * DH; e += 128) {
      int r = e >> 6, cc = e & 63;
      sT[r * STG_NK + cc] = (__bf16)rk_[(size_t)(c * 128 + r) * DH + cc];
    }
    __syncthreads();
#pragma unroll
    for (int nt = 0; nt < 4; ++nt) {
      const int n0 = colHalf * 64 + nt * 16;
      v8f acc = {};
      acc = wmma_bf16(aq0, load_B16(sT, n0, 0,  STG_NK), acc);
      acc = wmma_bf16(aq1, load_B16(sT, n0, 32, STG_NK), acc);
      const int hi = (lane & 16) ? 8 : 0;
      const int l  = c * 128 + n0 + (lane & 15);
#pragma unroll
      for (int r = 0; r < 8; ++r) {
        const int row = rowHalf * 16 + hi + r;
        const int qg  = qBase + row;
        float* rp = sS + row * S_LEN;
        if (l == 0) {
          rp[qg & (S_LEN - 1)] += acc[r];
        } else if (l < LMAX) {
          rp[(qg + l) & (S_LEN - 1)] += acc[r];
          rp[(qg - l) & (S_LEN - 1)] += acc[r];
        } else if (l == LMAX) {
          rp[(qg + LMAX) & (S_LEN - 1)] += acc[r];
        } // l > 512: discard (padding)
      }
    }
  }
  __syncthreads();

  // ---- Step 3: row softmax (fp32), write attn as bf16 (8 rows per wave)
  for (int rr = 0; rr < 8; ++rr) {
    const int row = wave * 8 + rr;
    float x[32];
    float m = -3.0e38f;
#pragma unroll
    for (int j = 0; j < 32; ++j) {
      x[j] = sS[row * S_LEN + lane + 32 * j];
      m = fmaxf(m, x[j]);
    }
#pragma unroll
    for (int off = 16; off > 0; off >>= 1) m = fmaxf(m, __shfl_xor(m, off, 32));
    float sum = 0.0f;
#pragma unroll
    for (int j = 0; j < 32; ++j) { x[j] = __expf(x[j] - m); sum += x[j]; }
#pragma unroll
    for (int off = 16; off > 0; off >>= 1) sum += __shfl_xor(sum, off, 32);
    const float inv = 1.0f / sum;
#pragma unroll
    for (int j = 0; j < 32; ++j)
      sA[row * SA_STRIDE + lane + 32 * j] = (__bf16)(x[j] * inv);
  }
  __syncthreads();

  // ---- Step 4: W[q,l] = attn[q,(q+l)%S] + attn[q,(q-l)%S]  (bf16, zero-padded)
  for (int e = tid; e < QT * WCOLS; e += 128) {
    const int row = e / WCOLS;
    const int l   = e - row * WCOLS;
    const int qg  = qBase + row;
    float wv = 0.0f;
    const __bf16* ap = sA + row * SA_STRIDE;
    if (l == 0)          wv = (float)ap[qg & (S_LEN - 1)];
    else if (l < LMAX)   wv = (float)ap[(qg + l) & (S_LEN - 1)] +
                              (float)ap[(qg - l) & (S_LEN - 1)];
    else if (l == LMAX)  wv = (float)ap[(qg + LMAX) & (S_LEN - 1)];
    sW[row * SW_STRIDE + l] = (__bf16)wv;
  }

  // ---- Step 5: out = attn @ V + W @ rel_v  (each wave: 16 rows x 32 d-cols)
  v8f o0 = {}, o1 = {};
  for (int c = 0; c < 8; ++c) {          // attn @ V, 128 keys per chunk
    __syncthreads();
    for (int e = tid; e < 128 * DH; e += 128) {
      int kk = e >> 6, dd = e & 63;
      sT[dd * STG_KN + kk] = (__bf16)vbh[(size_t)(c * 128 + kk) * DMODEL + dd];
    }
    __syncthreads();
#pragma unroll
    for (int ks = 0; ks < 4; ++ks) {
      v16bf a = load_A16(sA, rowHalf * 16, c * 128 + ks * 32, SA_STRIDE);
      o0 = wmma_bf16(a, load_B16(sT, colHalf * 32,      ks * 32, STG_KN), o0);
      o1 = wmma_bf16(a, load_B16(sT, colHalf * 32 + 16, ks * 32, STG_KN), o1);
    }
  }
  for (int c = 0; c < 5; ++c) {          // W @ rel_v, K = 4*128 + 32
    __syncthreads();
    const int rows = (c < 4) ? 128 : 32;
    for (int e = tid; e < rows * DH; e += 128) {
      int kk = e >> 6, dd = e & 63;
      sT[dd * STG_KN + kk] = (__bf16)rv_[(size_t)(c * 128 + kk) * DH + dd];
    }
    __syncthreads();
    const int nks = (c < 4) ? 4 : 1;
    for (int ks = 0; ks < nks; ++ks) {
      v16bf a = load_A16(sW, rowHalf * 16, c * 128 + ks * 32, SW_STRIDE);
      o0 = wmma_bf16(a, load_B16(sT, colHalf * 32,      ks * 32, STG_KN), o0);
      o1 = wmma_bf16(a, load_B16(sT, colHalf * 32 + 16, ks * 32, STG_KN), o1);
    }
  }

  // ---- Epilogue: fp32 store, C-layout (M = r + 8*laneHi, N = lane&15)
  {
    const int row0 = rowHalf * 16 + ((lane & 16) ? 8 : 0);
    const int d0   = colHalf * 32 + (lane & 15);
#pragma unroll
    for (int r = 0; r < 8; ++r) {
      const int qg = qBase + row0 + r;
      obh[(size_t)qg * DMODEL + d0]      = o0[r];
      obh[(size_t)qg * DMODEL + d0 + 16] = o1[r];
    }
  }
}

extern "C" void kernel_launch(void* const* d_in, const int* in_sizes, int n_in,
                              void* d_out, int out_size, void* d_ws, size_t ws_size,
                              hipStream_t stream) {
  const float* q  = (const float*)d_in[0];
  const float* k  = (const float*)d_in[1];
  const float* v  = (const float*)d_in[2];
  const float* rk = (const float*)d_in[3];
  const float* rv = (const float*)d_in[4];
  float* out = (float*)d_out;

  const int B = in_sizes[0] / (S_LEN * DMODEL);   // 4
  const int blocks = B * NHEADS * (S_LEN / QT);   // 2048

  (void)hipFuncSetAttribute((const void*)crpa_wmma_kernel,
                            hipFuncAttributeMaxDynamicSharedMemorySize,
                            SMEM_BYTES);
  crpa_wmma_kernel<<<blocks, 128, SMEM_BYTES, stream>>>(q, k, v, rk, rv, out);
}
